// MSAAttention_64707977281625
// MI455X (gfx1250) — matrix operand order repaired and checked by reading, not compiled, not run on hardware
//
#include <hip/hip_runtime.h>
#include <hip/hip_bf16.h>
#include <math.h>

#define DEV __device__ __forceinline__

#if defined(__has_builtin)
#if __has_builtin(__builtin_amdgcn_global_load_async_to_lds_b128)
#define USE_ASYNC_LDS 1
#endif
#endif

typedef __attribute__((ext_vector_type(4)))  __bf16 bf16x4;
typedef __attribute__((ext_vector_type(8)))  __bf16 bf16x8;
typedef __attribute__((ext_vector_type(16))) __bf16 bf16x16;
typedef __attribute__((ext_vector_type(8)))  float  f32x8;
typedef int v4i_gnu __attribute__((vector_size(16)));   // matches builtin param

constexpr int C_IN = 256, C_HID = 32, H = 8, C_Z = 128;
constexpr int N_SEQ = 128, N_RES = 384;
constexpr int M_TOT = N_SEQ * N_RES;         // 49152 rows for the big GEMMs

// ---------------------------------------------------------------- utilities

DEV float wave_sum32(float v) {
#pragma unroll
  for (int m = 16; m >= 1; m >>= 1) v += __shfl_xor(v, m, 32);
  return v;
}

#ifdef USE_ASYNC_LDS
// one lane-wide 16B async DMA: LDS[l] = MEM[g]   (tracked by ASYNCcnt)
// as(3) pointers are 32-bit; low 32 bits of a flat LDS address ARE the LDS
// offset, so the integer cast is exact.
DEV void async_copy16(const __bf16* g, __bf16* l) {
  __builtin_amdgcn_global_load_async_to_lds_b128(
      (v4i_gnu*)g,
      (__attribute__((address_space(3))) v4i_gnu*)(unsigned)(size_t)l,
      0, 0);
}
#endif

DEV void wait_async0() {
#if defined(__has_builtin) && __has_builtin(__builtin_amdgcn_s_wait_asynccnt)
  __builtin_amdgcn_s_wait_asynccnt(0);
#else
  asm volatile("s_wait_asynccnt 0x0" ::: "memory");
#endif
}

// A fragment: 16x32 bf16, row-major source, one row per lane (lane%16).
// ISA layout: lanes 0-15 hold K = 0..7 and 16..23; lanes 16-31 hold K = 8..15 and 24..31.
DEV bf16x16 load_frag_a(const __bf16* base, int row, int row_stride) {
  const int lane = threadIdx.x & 31;
  const int kb = (lane < 16) ? 0 : 8;
  const __bf16* p = base + (size_t)row * row_stride;
  bf16x8 lo = *reinterpret_cast<const bf16x8*>(p + kb);
  bf16x8 hi = *reinterpret_cast<const bf16x8*>(p + kb + 16);
  return __builtin_shufflevector(lo, hi, 0,1,2,3,4,5,6,7,8,9,10,11,12,13,14,15);
}

// B fragment: 32x16 bf16; source is B-TRANSPOSED (n-major, contraction contiguous).
// ISA layout: lanes 0-15 hold K = 0..7 and 16..23; lanes 16-31 hold K = 8..15 and 24..31; column n = lane%16.
DEV bf16x16 load_frag_b(const __bf16* baseT, int n, int row_stride) {
  const int lane = threadIdx.x & 31;
  const int kh = (lane < 16) ? 0 : 8;
  const __bf16* p = baseT + (size_t)n * row_stride + kh;
  bf16x8 lo = *reinterpret_cast<const bf16x8*>(p);
  bf16x8 hi = *reinterpret_cast<const bf16x8*>(p + 16);
  return __builtin_shufflevector(lo, hi, 0,1,2,3,4,5,6,7,8,9,10,11,12,13,14,15);
}

DEV f32x8 wmma_bf16(bf16x16 a, bf16x16 b, f32x8 c) {
  return __builtin_amdgcn_wmma_f32_16x16x32_bf16(
      /*neg_a=*/false, a, /*neg_b=*/false, b,
      /*c_mod=*/(short)0, c, /*reuse_a=*/false, /*reuse_b=*/false);
}

// ----------------------------------------------------- layernorm(m) -> bf16

__global__ __launch_bounds__(256)
void ln_m_kernel(const float* __restrict__ m, const float* __restrict__ w,
                 const float* __restrict__ b, __bf16* __restrict__ out) {
  const int lane = threadIdx.x & 31, wave = threadIdx.x >> 5;
  const size_t row = (size_t)blockIdx.x * 8 + wave;          // 49152 rows
  const float* x = m + row * C_IN + lane * 8;
  float4 v0 = *reinterpret_cast<const float4*>(x);
  float4 v1 = *reinterpret_cast<const float4*>(x + 4);
  float s = v0.x + v0.y + v0.z + v0.w + v1.x + v1.y + v1.z + v1.w;
  float s2 = v0.x*v0.x + v0.y*v0.y + v0.z*v0.z + v0.w*v0.w +
             v1.x*v1.x + v1.y*v1.y + v1.z*v1.z + v1.w*v1.w;
  s = wave_sum32(s); s2 = wave_sum32(s2);
  const float mu = s * (1.0f / C_IN);
  const float rs = rsqrtf(s2 * (1.0f / C_IN) - mu * mu + 1e-5f);
  float xv[8] = {v0.x, v0.y, v0.z, v0.w, v1.x, v1.y, v1.z, v1.w};
  bf16x8 o;
#pragma unroll
  for (int j = 0; j < 8; j++) {
    int c = lane * 8 + j;
    o[j] = (__bf16)((xv[j] - mu) * rs * w[c] + b[c]);
  }
  *reinterpret_cast<bf16x8*>(out + row * C_IN + lane * 8) = o;
}

// -------------------------------------- layernorm(z) * w_pair -> pair bias

__global__ __launch_bounds__(256)
void ln_z_pair_kernel(const float* __restrict__ z, const float* __restrict__ w,
                      const float* __restrict__ b, const float* __restrict__ wp,
                      float* __restrict__ pairb) {
  const int lane = threadIdx.x & 31, wave = threadIdx.x >> 5;
  const size_t p = (size_t)blockIdx.x * 8 + wave;            // 147456 pairs
  const int qi = (int)(p / N_RES), ki = (int)(p % N_RES);
  float4 v = *reinterpret_cast<const float4*>(z + p * C_Z + lane * 4);
  float s = v.x + v.y + v.z + v.w;
  float s2 = v.x*v.x + v.y*v.y + v.z*v.z + v.w*v.w;
  s = wave_sum32(s); s2 = wave_sum32(s2);
  const float mu = s * (1.0f / C_Z);
  const float rs = rsqrtf(s2 * (1.0f / C_Z) - mu * mu + 1e-5f);
  float xv[4] = {v.x, v.y, v.z, v.w};
  float zn[4];
#pragma unroll
  for (int j = 0; j < 4; j++) {
    int c = lane * 4 + j;
    zn[j] = (xv[j] - mu) * rs * w[c] + b[c];
  }
#pragma unroll
  for (int h = 0; h < H; h++) {
    float part = 0.f;
#pragma unroll
    for (int j = 0; j < 4; j++) part += zn[j] * wp[(lane * 4 + j) * H + h];
    part = wave_sum32(part);
    if (lane == 0) pairb[((size_t)h * N_RES + qi) * N_RES + ki] = part;
  }
}

// -------------------------------- fp32 [256][256] -> bf16 transposed [n][k]

__global__ __launch_bounds__(256)
void convert_transpose_256(const float* __restrict__ in, __bf16* __restrict__ out) {
  const int idx = blockIdx.x * 256 + threadIdx.x;   // 65536 threads
  const int i = idx >> 8, j = idx & 255;
  out[j * 256 + i] = (__bf16)in[idx];
}

// ----------------------------------------------------- WMMA GEMM + epilogue
// C[M,256] = A[M,256] @ B[256,256] ; Bt is B-transposed bf16 [n][k].
// Block tile 64(M) x 128(N), 8 waves in 2x4, wave tile 32x32 (4 WMMA/K-step),
// double-buffered LDS with async global->LDS DMA (ASYNCcnt) when available,
// one barrier per K-step, global_prefetch 2 steps ahead.
// MODE 0: Q (scale, bf16 [s][h][r][d])   MODE 1: K (bf16 [s][h][r][d])
// MODE 2: V (bf16 TRANSPOSED [s][h][d][r])
// MODE 3: G (sigmoid(x+bg), bf16 [s][h][r][d])
// MODE 4: OUT (x+bo, f32 [s][r][c])

template <int MODE>
__global__ __launch_bounds__(256)
void proj_gemm(const __bf16* __restrict__ A, const __bf16* __restrict__ Bt,
               const float* __restrict__ bias, void* __restrict__ outp,
               float scale) {
  __shared__ __bf16 As[2][64 * 32];
  __shared__ __bf16 Bs[2][128 * 32];
  const int tid = threadIdx.x, lane = tid & 31, wave = tid >> 5;
  const int wrow = wave >> 2, wcol = wave & 3;       // 2 x 4 wave grid
  const int m0 = blockIdx.x * 64, n0 = blockIdx.y * 128;

  const int ai = tid * 8;                            // A stage: 8 bf16 / thread
  const size_t a_off = (size_t)(m0 + (ai >> 5)) * 256 + (ai & 31);
  const int bi = tid * 16;                           // B stage: 16 bf16 / thread
  const size_t b_off = (size_t)(n0 + (bi >> 5)) * 256 + (bi & 31);

#ifdef USE_ASYNC_LDS
  // prologue: DMA chunk 0 straight into LDS buffer 0
  async_copy16(&A[a_off],      &As[0][ai]);
  async_copy16(&Bt[b_off],     &Bs[0][bi]);
  async_copy16(&Bt[b_off + 8], &Bs[0][bi + 8]);
#else
  bf16x8 sa, sb0, sb1;
  sa  = *reinterpret_cast<const bf16x8*>(&A[a_off]);
  sb0 = *reinterpret_cast<const bf16x8*>(&Bt[b_off]);
  sb1 = *reinterpret_cast<const bf16x8*>(&Bt[b_off + 8]);
  *reinterpret_cast<bf16x8*>(&As[0][ai])     = sa;
  *reinterpret_cast<bf16x8*>(&Bs[0][bi])     = sb0;
  *reinterpret_cast<bf16x8*>(&Bs[0][bi + 8]) = sb1;
#endif

  f32x8 acc[2][2] = {};
  for (int it = 0; it < 8; ++it) {
    const int cur = it & 1;
#ifdef USE_ASYNC_LDS
    wait_async0();                                   // my chunk `it` landed
    __syncthreads();                                 // everyone's landed
    if (it < 7) {                                    // DMA next chunk while we compute
      const int kc = (it + 1) * 32, nxt = cur ^ 1;
      async_copy16(&A[a_off + kc],      &As[nxt][ai]);
      async_copy16(&Bt[b_off + kc],     &Bs[nxt][bi]);
      async_copy16(&Bt[b_off + kc + 8], &Bs[nxt][bi + 8]);
    }
#else
    __syncthreads();
    if (it < 7) {                                    // stage next K-chunk via regs
      const int kc = (it + 1) * 32;
      sa  = *reinterpret_cast<const bf16x8*>(&A[a_off + kc]);
      sb0 = *reinterpret_cast<const bf16x8*>(&Bt[b_off + kc]);
      sb1 = *reinterpret_cast<const bf16x8*>(&Bt[b_off + kc + 8]);
    }
#endif
    if (it < 6) {                                    // global_prefetch_b8 path
      __builtin_prefetch(&A[a_off + (it + 2) * 32], 0, 0);
      __builtin_prefetch(&Bt[b_off + (it + 2) * 32], 0, 0);
    }
    bf16x16 af0 = load_frag_a(As[cur], wrow * 32 + (lane & 15), 32);
    bf16x16 af1 = load_frag_a(As[cur], wrow * 32 + 16 + (lane & 15), 32);
    bf16x16 bf0 = load_frag_b(Bs[cur], wcol * 32 + (lane & 15), 32);
    bf16x16 bf1 = load_frag_b(Bs[cur], wcol * 32 + 16 + (lane & 15), 32);
    acc[0][0] = wmma_bf16(af0, bf0, acc[0][0]);
    acc[0][1] = wmma_bf16(af0, bf1, acc[0][1]);
    acc[1][0] = wmma_bf16(af1, bf0, acc[1][0]);
    acc[1][1] = wmma_bf16(af1, bf1, acc[1][1]);
#ifndef USE_ASYNC_LDS
    if (it < 7) {
      const int nxt = cur ^ 1;
      *reinterpret_cast<bf16x8*>(&As[nxt][ai])     = sa;
      *reinterpret_cast<bf16x8*>(&Bs[nxt][bi])     = sb0;
      *reinterpret_cast<bf16x8*>(&Bs[nxt][bi + 8]) = sb1;
    }
#endif
  }

#pragma unroll
  for (int mi = 0; mi < 2; mi++) {
#pragma unroll
    for (int ni = 0; ni < 2; ni++) {
      const int col = n0 + wcol * 32 + ni * 16 + (lane & 15);
      const int h = col >> 5, d = col & 31;
#pragma unroll
      for (int i = 0; i < 8; i++) {
        const int grow = m0 + wrow * 32 + mi * 16 + i + ((lane < 16) ? 0 : 8);
        const int s = grow / N_RES, r = grow % N_RES;
        float v = acc[mi][ni][i];
        if (MODE == 0) {
          ((__bf16*)outp)[(((size_t)(s * H + h)) * N_RES + r) * C_HID + d] = (__bf16)(v * scale);
        } else if (MODE == 1) {
          ((__bf16*)outp)[(((size_t)(s * H + h)) * N_RES + r) * C_HID + d] = (__bf16)v;
        } else if (MODE == 2) {
          ((__bf16*)outp)[(((size_t)(s * H + h)) * C_HID + d) * N_RES + r] = (__bf16)v;
        } else if (MODE == 3) {
          v += bias[col];
          ((__bf16*)outp)[(((size_t)(s * H + h)) * N_RES + r) * C_HID + d] =
              (__bf16)(1.0f / (1.0f + expf(-v)));
        } else {
          ((float*)outp)[(size_t)grow * 256 + col] = v + bias[col];
        }
      }
    }
  }
}

// ------------------------------------------------- fused attention per (s,h)
// grid: (24 q-tiles, 8 heads, 128 seqs), block: 256 threads / 8 waves.

__global__ __launch_bounds__(256)
void attn_kernel(const __bf16* __restrict__ q, const __bf16* __restrict__ k,
                 const __bf16* __restrict__ vT, const __bf16* __restrict__ g,
                 const float* __restrict__ pairb, const float* __restrict__ mask,
                 __bf16* __restrict__ og) {
  __shared__ float  lg[16 * N_RES];     // logits f32      (24 KB)
  __shared__ __bf16 at[16 * N_RES];     // softmax bf16    (12 KB)
  __shared__ float  osum[16 * C_HID];   // AV reduction    ( 2 KB)

  const int tid = threadIdx.x, lane = tid & 31, wave = tid >> 5;
  const int q0 = blockIdx.x * 16, h = blockIdx.y, s = blockIdx.z;
  const size_t sh = (size_t)s * H + h;

  osum[tid] = 0.f; osum[tid + 256] = 0.f;

  // ---- phase 1: logits = Q·K^T + pair_bias + mask_bias
  const __bf16* qbase = q + (sh * N_RES + q0) * C_HID;
  bf16x16 afrag = load_frag_a(qbase, lane & 15, C_HID);
#pragma unroll
  for (int t = 0; t < 3; t++) {
    const int kt = wave + 8 * t;                       // 24 k-tiles of 16
    const __bf16* kbase = k + (sh * N_RES + kt * 16) * C_HID;
    bf16x16 bfrag = load_frag_b(kbase, lane & 15, C_HID);
    f32x8 c = {};
    c = wmma_bf16(afrag, bfrag, c);
    const int col = kt * 16 + (lane & 15);
    const float mb = 1e9f * (mask[(size_t)s * N_RES + col] - 1.0f);
#pragma unroll
    for (int i = 0; i < 8; i++) {
      const int row = i + ((lane < 16) ? 0 : 8);
      lg[row * N_RES + col] =
          c[i] + mb + pairb[((size_t)h * N_RES + (q0 + row)) * N_RES + col];
    }
  }
  __syncthreads();

  // ---- phase 2: row softmax (16 threads per row, 24 cols each)
  {
    const int row = tid >> 4, g0 = tid & 15;
    float mx = -3.0e38f;
#pragma unroll
    for (int j = 0; j < 24; j++) mx = fmaxf(mx, lg[row * N_RES + g0 + 16 * j]);
#pragma unroll
    for (int m = 1; m <= 8; m <<= 1) mx = fmaxf(mx, __shfl_xor(mx, m, 32));
    float sum = 0.f;
#pragma unroll
    for (int j = 0; j < 24; j++) sum += expf(lg[row * N_RES + g0 + 16 * j] - mx);
#pragma unroll
    for (int m = 1; m <= 8; m <<= 1) sum += __shfl_xor(sum, m, 32);
    const float inv = 1.0f / sum;
#pragma unroll
    for (int j = 0; j < 24; j++) {
      const int c = g0 + 16 * j;
      at[row * N_RES + c] = (__bf16)(expf(lg[row * N_RES + c] - mx) * inv);
    }
  }
  __syncthreads();

  // ---- phase 3: O = A·V  (wave -> (d-halftile, K-quarter), LDS reduce)
  {
    const int ntile = wave & 1, kq = wave >> 1;
    f32x8 cp = {};
#pragma unroll
    for (int t = 0; t < 3; t++) {
      const int step = kq + 4 * t;                     // 12 K-steps of 32
      bf16x16 a2 = load_frag_a(at + step * 32, lane & 15, N_RES);
      const __bf16* vbase = vT + (sh * C_HID + ntile * 16) * N_RES + step * 32;
      bf16x16 b2 = load_frag_b(vbase, lane & 15, N_RES);
      cp = wmma_bf16(a2, b2, cp);
    }
#pragma unroll
    for (int i = 0; i < 8; i++) {
      const int row = i + ((lane < 16) ? 0 : 8);
      const int d = ntile * 16 + (lane & 15);
      atomicAdd(&osum[row * C_HID + d], cp[i]);
    }
  }
  __syncthreads();

  // ---- phase 4: gate + store og bf16 [s][r][h*32+d]
#pragma unroll
  for (int rep = 0; rep < 2; rep++) {
    const int idx = tid + rep * 256;
    const int row = idx >> 5, d = idx & 31;
    const float gv = (float)g[(sh * N_RES + (q0 + row)) * C_HID + d];
    og[((size_t)s * N_RES + (q0 + row)) * 256 + h * C_HID + d] =
        (__bf16)(osum[idx] * gv);
  }
}

// ------------------------------------------------------------------- launch

extern "C" void kernel_launch(void* const* d_in, const int* in_sizes, int n_in,
                              void* d_out, int out_size, void* d_ws, size_t ws_size,
                              hipStream_t stream) {
  (void)in_sizes; (void)n_in; (void)out_size; (void)ws_size;
  const float* m      = (const float*)d_in[0];
  const float* z      = (const float*)d_in[1];
  const float* mask   = (const float*)d_in[2];
  const float* ln_m_w = (const float*)d_in[3];
  const float* ln_m_b = (const float*)d_in[4];
  const float* ln_z_w = (const float*)d_in[5];
  const float* ln_z_b = (const float*)d_in[6];
  const float* w_pair = (const float*)d_in[7];
  const float* wq     = (const float*)d_in[8];
  const float* wk     = (const float*)d_in[9];
  const float* wv     = (const float*)d_in[10];
  const float* wg     = (const float*)d_in[11];
  const float* bg     = (const float*)d_in[12];
  const float* wo     = (const float*)d_in[13];
  const float* bo     = (const float*)d_in[14];

  char* ws = (char*)d_ws;
  constexpr size_t SZ_MLN  = (size_t)M_TOT * C_IN * 2;                 // 25.2 MB
  constexpr size_t SZ_HEAD = (size_t)N_SEQ * H * N_RES * C_HID * 2;    // 25.2 MB
  constexpr size_t SZ_PB   = (size_t)H * N_RES * N_RES * 4;            // 4.7 MB
  constexpr size_t SZ_W    = (size_t)256 * 256 * 2;

  __bf16* mln   = (__bf16*)(ws);
  __bf16* qb    = (__bf16*)(ws + SZ_MLN);
  __bf16* kb    = (__bf16*)(ws + SZ_MLN + SZ_HEAD);
  __bf16* vTb   = (__bf16*)(ws + SZ_MLN + 2 * SZ_HEAD);
  __bf16* gb    = (__bf16*)(ws + SZ_MLN + 3 * SZ_HEAD);
  float*  pb    = (float*) (ws + SZ_MLN + 4 * SZ_HEAD);
  __bf16* wqT   = (__bf16*)(ws + SZ_MLN + 4 * SZ_HEAD + SZ_PB);
  __bf16* wkT   = (__bf16*)(ws + SZ_MLN + 4 * SZ_HEAD + SZ_PB + 1 * SZ_W);
  __bf16* wvT   = (__bf16*)(ws + SZ_MLN + 4 * SZ_HEAD + SZ_PB + 2 * SZ_W);
  __bf16* wgT   = (__bf16*)(ws + SZ_MLN + 4 * SZ_HEAD + SZ_PB + 3 * SZ_W);
  __bf16* woT   = (__bf16*)(ws + SZ_MLN + 4 * SZ_HEAD + SZ_PB + 4 * SZ_W);
  __bf16* og    = mln;   // m_ln dead after projections; reuse the buffer

  ln_m_kernel<<<M_TOT / 8, 256, 0, stream>>>(m, ln_m_w, ln_m_b, mln);
  ln_z_pair_kernel<<<(N_RES * N_RES) / 8, 256, 0, stream>>>(z, ln_z_w, ln_z_b, w_pair, pb);

  convert_transpose_256<<<256, 256, 0, stream>>>(wq, wqT);
  convert_transpose_256<<<256, 256, 0, stream>>>(wk, wkT);
  convert_transpose_256<<<256, 256, 0, stream>>>(wv, wvT);
  convert_transpose_256<<<256, 256, 0, stream>>>(wg, wgT);
  convert_transpose_256<<<256, 256, 0, stream>>>(wo, woT);

  const dim3 gGemm(M_TOT / 64, 2);                   // 64x128 block tiles
  const float scale = 0.17677669529663689f;          // 32^-0.5
  proj_gemm<0><<<gGemm, 256, 0, stream>>>(mln, wqT, nullptr, qb, scale);
  proj_gemm<1><<<gGemm, 256, 0, stream>>>(mln, wkT, nullptr, kb, 1.0f);
  proj_gemm<2><<<gGemm, 256, 0, stream>>>(mln, wvT, nullptr, vTb, 1.0f);
  proj_gemm<3><<<gGemm, 256, 0, stream>>>(mln, wgT, bg, gb, 1.0f);

  attn_kernel<<<dim3(N_RES / 16, H, N_SEQ), 256, 0, stream>>>(qb, kb, vTb, gb, pb, mask, og);

  proj_gemm<4><<<gGemm, 256, 0, stream>>>(og, woT, bo, d_out, 1.0f);
}
